// MixtureOfExperts_75737453297724
// MI455X (gfx1250) — matrix operand order repaired and checked by reading, not compiled, not run on hardware
//
#include <hip/hip_runtime.h>

// ---------------------------------------------------------------------------
// MoE (top-2 of 8) for MI455X / gfx1250, wave32 + v_wmma_f32_16x16x32_bf16.
// gate+route -> bf16 casts (weights pre-transposed to [n][k]) -> per-expert
// (x@W1+b1, relu) -> (@W2+b2) scaled-accumulate into fp32 out.
// GEMMs: 128x128x64 tiles, 8 waves, DOUBLE-BUFFERED LDS fed by CDNA5
// GLOBAL_LOAD_ASYNC_TO_LDS_B128 (ASYNCcnt-paced), zero-VGPR staging.
// Fragment loads are explicit b128 vectors so DS offsets fold as immediates
// (avoids 2addr_b32 selection + WMMA->VALU WAR v_nops).
// ---------------------------------------------------------------------------

typedef __attribute__((ext_vector_type(16))) __bf16 v16bf;
typedef __attribute__((ext_vector_type(8)))  float  v8f;
typedef __attribute__((ext_vector_type(4)))  __bf16 bf16x4;
typedef __attribute__((ext_vector_type(4)))  int    v4i;

#define T_TOK   16384   // B*S
#define DIM     1024    // D
#define HID     2048    // H
#define NEXP    8       // E

#define BM 128
#define BN 128
#define BK 64
#define LDK (BK + 8)            // 72 bf16 = 144B rows; 16B aligned chunks
#define THREADS 256             // 8 wave32
#define ST_ELEMS ((BM + BN) * LDK)          // bf16 per stage (A tile + B tile)
#define SMEM_BYTES (2 * ST_ELEMS * 2)       // two stages, 73728 B (dynamic LDS)

#if __has_builtin(__builtin_amdgcn_global_load_async_to_lds_b128)
#define HAVE_ASYNC_LDS 1
typedef __attribute__((address_space(1))) v4i* v4i_gp;   // global (AS1)
typedef __attribute__((address_space(3))) v4i* v4i_lp;   // LDS (AS3, 32-bit)
__device__ __forceinline__ void async_b128(const __bf16* g, __bf16* l) {
  // AS-qualified pointers built via integer casts (always-legal conversions).
  // Low 32 bits of a generic LDS pointer are the LDS byte address (ISA 10.2).
  __builtin_amdgcn_global_load_async_to_lds_b128(
      (v4i_gp)(unsigned long long)(uintptr_t)g,
      (v4i_lp)(unsigned int)(uintptr_t)l,
      0, 0);
}
#else
#define HAVE_ASYNC_LDS 0
#endif

#if __has_builtin(__builtin_amdgcn_s_wait_asynccnt)
#define WAIT_ASYNC_LE8() __builtin_amdgcn_s_wait_asynccnt(8)
#define WAIT_ASYNC_0()   __builtin_amdgcn_s_wait_asynccnt(0)
#else
#define WAIT_ASYNC_LE8() asm volatile("s_wait_asynccnt 0x8" ::: "memory")
#define WAIT_ASYNC_0()   asm volatile("s_wait_asynccnt 0x0" ::: "memory")
#endif

// A fragment (16x32 bf16, M=lane&15): lane<16 holds K {kb..kb+7, kb+16..kb+23}
// with kb=0; lanes>=16 kb=8.  Two explicit b128 LDS loads (16B aligned).
__device__ __forceinline__ v16bf frag_ld_a(const __bf16* row, int kb) {
  v16bf f;
  uint4* q = reinterpret_cast<uint4*>(&f);
  q[0] = *reinterpret_cast<const uint4*>(row + kb);
  q[1] = *reinterpret_cast<const uint4*>(row + kb + 16);
  return f;
}

// B fragment (32x16 bf16, N=lane&15): lane<16 holds K {0..7, 16..23}, lanes>=16
// K {8..15, 24..31}.  `p` -> two b128 LDS loads at p and p+16.
__device__ __forceinline__ v16bf frag_ld_b(const __bf16* p) {
  v16bf f;
  uint4* q = reinterpret_cast<uint4*>(&f);
  q[0] = *reinterpret_cast<const uint4*>(p);
  q[1] = *reinterpret_cast<const uint4*>(p + 16);
  return f;
}

// One BK=64 k-stage of the 128x128 tile: 2 k-steps x 8 n-tiles of WMMA.
__device__ __forceinline__ void mm_stage(const __bf16* as, const __bf16* bs,
                                         int wave, int lm, int kb, int kh,
                                         v8f acc[8]) {
#pragma unroll
  for (int kk = 0; kk < BK; kk += 32) {
    const v16bf af = frag_ld_a(as + (wave * 16 + lm) * LDK + kk, kb);
#pragma unroll
    for (int nt = 0; nt < 8; ++nt) {
      const v16bf bf = frag_ld_b(bs + (nt * 16 + lm) * LDK + kk + kb);
      acc[nt] = __builtin_amdgcn_wmma_f32_16x16x32_bf16(
          false, af, false, bf, (short)0, acc[nt], false, false);
    }
  }
}

// ---------------------------------------------------------------------------
// Gating: one wave per token; butterfly reduce 8 logits; top-2 (jax tie-break),
// renormalized weights; atomic append to per-expert compacted lists.
// ---------------------------------------------------------------------------
__global__ __launch_bounds__(256)
void moe_gate(const float* __restrict__ x, const float* __restrict__ Wg,
              const float* __restrict__ bg, int* __restrict__ cnt,
              int* __restrict__ tok, float* __restrict__ tokw)
{
  const int lane = threadIdx.x & 31;
  const int wave = threadIdx.x >> 5;
  const int t = blockIdx.x * 8 + wave;

  float a[NEXP];
#pragma unroll
  for (int e = 0; e < NEXP; ++e) a[e] = 0.f;

  const float* xr = x + (size_t)t * DIM;
  for (int d = lane; d < DIM; d += 32) {
    const float xv = xr[d];
    const float4 w0 = *reinterpret_cast<const float4*>(Wg + (size_t)d * NEXP);
    const float4 w1 = *reinterpret_cast<const float4*>(Wg + (size_t)d * NEXP + 4);
    a[0] += xv * w0.x; a[1] += xv * w0.y; a[2] += xv * w0.z; a[3] += xv * w0.w;
    a[4] += xv * w1.x; a[5] += xv * w1.y; a[6] += xv * w1.z; a[7] += xv * w1.w;
  }
#pragma unroll
  for (int off = 16; off > 0; off >>= 1) {
#pragma unroll
    for (int e = 0; e < NEXP; ++e) a[e] += __shfl_xor(a[e], off, 32);
  }

  if (lane == 0) {
    float l[NEXP];
#pragma unroll
    for (int e = 0; e < NEXP; ++e) l[e] = a[e] + bg[e];
    int i0 = 0;
#pragma unroll
    for (int e = 1; e < NEXP; ++e) if (l[e] > l[i0]) i0 = e;
    int i1 = (i0 == 0) ? 1 : 0;
#pragma unroll
    for (int e = 0; e < NEXP; ++e)
      if (e != i0 && e != i1 && l[e] > l[i1]) i1 = e;
    const float r  = __expf(l[i1] - l[i0]);   // <= 1
    const float w0 = 1.f / (1.f + r);
    const float w1 = r * w0;
    const int p0 = atomicAdd(cnt + i0, 1);
    tok[i0 * T_TOK + p0]  = t;
    tokw[i0 * T_TOK + p0] = w0;
    const int p1 = atomicAdd(cnt + i1, 1);
    tok[i1 * T_TOK + p1]  = t;
    tokw[i1 * T_TOK + p1] = w1;
  }
}

// ---------------------------------------------------------------------------
// fp32 -> bf16 straight cast (x activations)
// ---------------------------------------------------------------------------
__global__ __launch_bounds__(256)
void cast_f32_bf16(const float* __restrict__ in, __bf16* __restrict__ out, long n)
{
  const long i = ((long)blockIdx.x * 256 + threadIdx.x) * 4;
  if (i >= n) return;
  const float4 v = *reinterpret_cast<const float4*>(in + i);
  bf16x4 o;
  o.x = (__bf16)v.x; o.y = (__bf16)v.y; o.z = (__bf16)v.z; o.w = (__bf16)v.w;
  *reinterpret_cast<bf16x4*>(out + i) = o;
}

// ---------------------------------------------------------------------------
// fp32 [e][K][N] -> bf16 [e][N][K] tiled transpose-cast (weights).
// ---------------------------------------------------------------------------
__global__ __launch_bounds__(256)
void transpose_cast(const float* __restrict__ in, __bf16* __restrict__ outT,
                    int K, int N)
{
  __shared__ float tile[32][33];
  const int e = blockIdx.z;
  const float* src = in + (size_t)e * K * N;
  __bf16* dst = outT + (size_t)e * N * K;
  const int n0 = blockIdx.x * 32;
  const int k0 = blockIdx.y * 32;
  const int tx = threadIdx.x & 31;
  const int ty = threadIdx.x >> 5;
#pragma unroll
  for (int i = 0; i < 32; i += 8)
    tile[ty + i][tx] = src[(size_t)(k0 + ty + i) * N + (n0 + tx)];
  __syncthreads();
#pragma unroll
  for (int i = 0; i < 32; i += 8)
    dst[(size_t)(n0 + ty + i) * K + (k0 + tx)] = (__bf16)tile[tx][ty + i];
}

// ---------------------------------------------------------------------------
// FFN1: h[rows, n0:n0+128] = relu(gather(x)[rows] @ W1[e] + b1[e]) (bf16 out)
// w1t is [H][D] (pre-transposed).
// ---------------------------------------------------------------------------
__global__ __launch_bounds__(THREADS, 2)
void moe_ffn1(const __bf16* __restrict__ xb, const __bf16* __restrict__ w1t,
              const float* __restrict__ b1, const int* __restrict__ tokp,
              const int* __restrict__ cntp, __bf16* __restrict__ hws)
{
  const int cnt  = *cntp;
  const int row0 = blockIdx.x * BM;
  if (row0 >= cnt) return;
  const int n0 = blockIdx.y * BN;

  extern __shared__ __bf16 smem[];   // [2][ A: BM*LDK | B: BN*LDK ]

  const int tid  = threadIdx.x;
  const int lane = tid & 31;
  const int wave = tid >> 5;
  const int lm   = lane & 15;
  const int kb   = (lane >> 4) * 8;    // A frag k-base
  const int kh   = (lane >> 4) * 16;   // unused; B frag also uses kb

  // Per-thread staging coords: 4 b128 chunks each for A and B per stage.
  int loff[4], aoff[4], boff[4];
#pragma unroll
  for (int j = 0; j < 4; ++j) {
    const int c  = tid + j * THREADS;          // 0..1023
    const int r  = c >> 3;
    const int kc = (c & 7) * 8;
    loff[j] = r * LDK + kc;
    const int gr = row0 + r;
    const int t  = (gr < cnt) ? tokp[gr] : 0;
    aoff[j] = t * DIM + kc;
    boff[j] = (n0 + r) * DIM + kc;
  }

  v8f acc[8] = {};

#if HAVE_ASYNC_LDS
  // Prologue: stage 0 -> buf0, stage 1 -> buf1 (8 async b128 per wave each).
#pragma unroll
  for (int j = 0; j < 4; ++j) {
    async_b128(xb  + aoff[j], smem + loff[j]);
    async_b128(w1t + boff[j], smem + BM * LDK + loff[j]);
  }
#pragma unroll
  for (int j = 0; j < 4; ++j) {
    async_b128(xb  + aoff[j] + BK, smem + ST_ELEMS + loff[j]);
    async_b128(w1t + boff[j] + BK, smem + ST_ELEMS + BM * LDK + loff[j]);
  }
  for (int k0 = 0; k0 < DIM; k0 += BK) {
    __bf16* as = smem + ((k0 / BK) & 1) * ST_ELEMS;
    __bf16* bs = as + BM * LDK;
    if (k0 + BK < DIM) WAIT_ASYNC_LE8(); else WAIT_ASYNC_0();
    __syncthreads();                       // stage k0 fully in LDS (all waves)
    mm_stage(as, bs, wave, lm, kb, kh, acc);
    __syncthreads();                       // everyone done reading this buffer
    if (k0 + 2 * BK < DIM) {               // refill it with stage k0+2
#pragma unroll
      for (int j = 0; j < 4; ++j) {
        async_b128(xb  + aoff[j] + k0 + 2 * BK, as + loff[j]);
        async_b128(w1t + boff[j] + k0 + 2 * BK, bs + loff[j]);
      }
    }
  }
#else
  // Fallback: synchronous single-buffer staging.
  __bf16* as = smem;
  __bf16* bs = smem + BM * LDK;
  for (int k0 = 0; k0 < DIM; k0 += BK) {
    __syncthreads();
#pragma unroll
    for (int j = 0; j < 4; ++j) {
      *reinterpret_cast<uint4*>(as + loff[j]) =
          *reinterpret_cast<const uint4*>(xb + aoff[j] + k0);
      *reinterpret_cast<uint4*>(bs + loff[j]) =
          *reinterpret_cast<const uint4*>(w1t + boff[j] + k0);
    }
    __syncthreads();
    mm_stage(as, bs, wave, lm, kb, kh, acc);
  }
#endif

  // Epilogue: +bias, relu, bf16 store. C layout: VGPR r -> M = r + 8*(lane>=16).
  const int mo = (lane >> 4) * 8;
#pragma unroll
  for (int nt = 0; nt < 8; ++nt) {
    const int ncol   = n0 + nt * 16 + lm;
    const float bias = b1[ncol];
#pragma unroll
    for (int r = 0; r < 8; ++r) {
      const int grow = row0 + wave * 16 + mo + r;
      if (grow < cnt) {
        float v = acc[nt][r] + bias;
        v = v > 0.f ? v : 0.f;
        hws[(size_t)grow * HID + ncol] = (__bf16)v;
      }
    }
  }
}

// ---------------------------------------------------------------------------
// FFN2: out[t, n0:n0+128] += w_t * (h[rows] @ W2[e] + b2[e]).  w2t is [D][H].
// Plain RMW is safe: rows/cols unique within a launch, launches serialize.
// ---------------------------------------------------------------------------
__global__ __launch_bounds__(THREADS, 2)
void moe_ffn2(const __bf16* __restrict__ hws, const __bf16* __restrict__ w2t,
              const float* __restrict__ b2, const int* __restrict__ tokp,
              const float* __restrict__ tokwp, const int* __restrict__ cntp,
              float* __restrict__ out)
{
  const int cnt  = *cntp;
  const int row0 = blockIdx.x * BM;
  if (row0 >= cnt) return;
  const int n0 = blockIdx.y * BN;

  extern __shared__ __bf16 smem[];

  const int tid  = threadIdx.x;
  const int lane = tid & 31;
  const int wave = tid >> 5;
  const int lm   = lane & 15;
  const int kb   = (lane >> 4) * 8;
  const int kh   = (lane >> 4) * 16;

  int loff[4], aoff[4], boff[4];
#pragma unroll
  for (int j = 0; j < 4; ++j) {
    const int c  = tid + j * THREADS;
    const int r  = c >> 3;
    const int kc = (c & 7) * 8;
    loff[j] = r * LDK + kc;
    aoff[j] = (row0 + r) * HID + kc;   // rows always in-bounds (<T)
    boff[j] = (n0 + r) * HID + kc;
  }

  v8f acc[8] = {};

#if HAVE_ASYNC_LDS
#pragma unroll
  for (int j = 0; j < 4; ++j) {
    async_b128(hws + aoff[j], smem + loff[j]);
    async_b128(w2t + boff[j], smem + BM * LDK + loff[j]);
  }
#pragma unroll
  for (int j = 0; j < 4; ++j) {
    async_b128(hws + aoff[j] + BK, smem + ST_ELEMS + loff[j]);
    async_b128(w2t + boff[j] + BK, smem + ST_ELEMS + BM * LDK + loff[j]);
  }
  for (int k0 = 0; k0 < HID; k0 += BK) {
    __bf16* as = smem + ((k0 / BK) & 1) * ST_ELEMS;
    __bf16* bs = as + BM * LDK;
    if (k0 + BK < HID) WAIT_ASYNC_LE8(); else WAIT_ASYNC_0();
    __syncthreads();
    mm_stage(as, bs, wave, lm, kb, kh, acc);
    __syncthreads();
    if (k0 + 2 * BK < HID) {
#pragma unroll
      for (int j = 0; j < 4; ++j) {
        async_b128(hws + aoff[j] + k0 + 2 * BK, as + loff[j]);
        async_b128(w2t + boff[j] + k0 + 2 * BK, bs + loff[j]);
      }
    }
  }
#else
  __bf16* as = smem;
  __bf16* bs = smem + BM * LDK;
  for (int k0 = 0; k0 < HID; k0 += BK) {
    __syncthreads();
#pragma unroll
    for (int j = 0; j < 4; ++j) {
      *reinterpret_cast<uint4*>(as + loff[j]) =
          *reinterpret_cast<const uint4*>(hws + aoff[j] + k0);
      *reinterpret_cast<uint4*>(bs + loff[j]) =
          *reinterpret_cast<const uint4*>(w2t + boff[j] + k0);
    }
    __syncthreads();
    mm_stage(as, bs, wave, lm, kb, kh, acc);
  }
#endif

  const int mo = (lane >> 4) * 8;
#pragma unroll
  for (int r = 0; r < 8; ++r) {
    const int grow = row0 + wave * 16 + mo + r;
    if (grow >= cnt) continue;
    const int   t  = tokp[grow];
    const float wt = tokwp[grow];
    float* orow = out + (size_t)t * DIM;
#pragma unroll
    for (int nt = 0; nt < 8; ++nt) {
      const int ncol = n0 + nt * 16 + lm;
      orow[ncol] += wt * (acc[nt][r] + b2[ncol]);
    }
  }
}

// ---------------------------------------------------------------------------
extern "C" void kernel_launch(void* const* d_in, const int* in_sizes, int n_in,
                              void* d_out, int out_size, void* d_ws, size_t ws_size,
                              hipStream_t stream)
{
  const float* x  = (const float*)d_in[0];
  const float* Wg = (const float*)d_in[1];
  const float* bg = (const float*)d_in[2];
  const float* W1 = (const float*)d_in[3];
  const float* b1 = (const float*)d_in[4];
  const float* W2 = (const float*)d_in[5];
  const float* b2 = (const float*)d_in[6];
  float* out = (float*)d_out;

  size_t off = 0;
  auto carve = [&](size_t bytes) -> char* {
    char* p = (char*)d_ws + off;
    off += (bytes + 255) & ~(size_t)255;
    return p;
  };
  __bf16* xb  = (__bf16*)carve((size_t)T_TOK * DIM * 2);        // 32 MB
  __bf16* w1t = (__bf16*)carve((size_t)NEXP * DIM * HID * 2);   // 32 MB  [e][H][D]
  __bf16* w2t = (__bf16*)carve((size_t)NEXP * HID * DIM * 2);   // 32 MB  [e][D][H]
  __bf16* hws = (__bf16*)carve((size_t)T_TOK * HID * 2);        // 64 MB
  int*    cnt = (int*)carve(NEXP * sizeof(int));
  int*    tok = (int*)carve((size_t)NEXP * T_TOK * sizeof(int));
  float*  tkw = (float*)carve((size_t)NEXP * T_TOK * sizeof(float));

  (void)hipMemsetAsync(cnt, 0, NEXP * sizeof(int), stream);
  (void)hipMemsetAsync(out, 0, (size_t)out_size * sizeof(float), stream);

  { const long n = (long)T_TOK * DIM;
    cast_f32_bf16<<<(n / 4 + 255) / 256, 256, 0, stream>>>(x, xb, n); }
  // W1 [e][D][H] -> w1t [e][H][D];  W2 [e][H][D] -> w2t [e][D][H]
  transpose_cast<<<dim3(HID / 32, DIM / 32, NEXP), 256, 0, stream>>>(W1, w1t, DIM, HID);
  transpose_cast<<<dim3(DIM / 32, HID / 32, NEXP), 256, 0, stream>>>(W2, w2t, HID, DIM);

  moe_gate<<<T_TOK / 8, 256, 0, stream>>>(x, Wg, bg, cnt, tok, tkw);

  for (int e = 0; e < NEXP; ++e) {
    moe_ffn1<<<dim3(T_TOK / BM, HID / BN), THREADS, SMEM_BYTES, stream>>>(
        xb, w1t + (size_t)e * DIM * HID, b1 + (size_t)e * HID,
        tok + (size_t)e * T_TOK, cnt + e, hws);
    moe_ffn2<<<dim3(T_TOK / BM, DIM / BN), THREADS, SMEM_BYTES, stream>>>(
        hws, w2t + (size_t)e * HID * DIM, b2 + (size_t)e * DIM,
        tok + (size_t)e * T_TOK, tkw + (size_t)e * T_TOK, cnt + e, out);
  }
}
